// SelfAttention_7834020348210
// MI455X (gfx1250) — matrix operand order repaired and checked by reading, not compiled, not run on hardware
//
#include <hip/hip_runtime.h>
#include <hip/hip_bf16.h>

// Fused causal self-attention for B=4, S=4096, D=256 (fp32 in/out).
// Kernel 1: QKV projection via v_wmma_f32_16x16x32_f16; Q,K row-major f16,
//           V stored TRANSPOSED ([B][D][S]) so the TDM can stage both K and V
//           blocks as plain 2D tiles.
// Kernel 2: flash attention; K/V blocks staged in LDS by tensor_load_to_lds
//           (TDM), DOUBLE-BUFFERED: block j+1's DMA overlaps block j's WMMAs.

#define B_ 4
#define S_ 4096
#define D_ 256
#define SCALE_ 0.0625f   // 1/sqrt(256)

typedef _Float16 v16h __attribute__((ext_vector_type(16)));
typedef _Float16 v8h  __attribute__((ext_vector_type(8)));
typedef float    v8f  __attribute__((ext_vector_type(8)));
typedef unsigned int u32x4 __attribute__((ext_vector_type(4)));
typedef int i32x4 __attribute__((ext_vector_type(4)));
typedef int i32x8 __attribute__((ext_vector_type(8)));

#define KT_STRIDE 272   // halves per K-block row: 256 + 8dw TDM pad (codes 6/7)
#define VT_STRIDE 72    // halves per V^T row: 64 + 4dw TDM pad (codes 4/3)
#define SC_STRIDE 72    // floats per score row (64 + 8 pad)
#define SET_HALVES (64 * KT_STRIDE + 256 * VT_STRIDE)  // one K+V buffer set

// The gfx1250-specific builtins only exist in the DEVICE compilation pass;
// the host pass parses this TU too, so gate everything on __HIP_DEVICE_COMPILE__.
#if defined(__HIP_DEVICE_COMPILE__)
#if !__has_builtin(__builtin_amdgcn_tensor_load_to_lds)
#error "CDNA5 TDM builtin __builtin_amdgcn_tensor_load_to_lds not available"
#endif
#if !__has_builtin(__builtin_amdgcn_s_wait_tensorcnt)
#error "CDNA5 builtin __builtin_amdgcn_s_wait_tensorcnt not available"
#endif
#define WAIT_TENSORCNT(n) __builtin_amdgcn_s_wait_tensorcnt(n)
#else
#define WAIT_TENSORCNT(n) ((void)0)
#endif

// Fragment layout (ISA 7.12.2), wave32:
//   A 16x32 f16: lane row M=lane%16; elem e -> K = 16*(lane>=16) + e (this code's order, same as B)
//   B 32x16 f16: lane col N=lane%16; elem e -> K = 16*(lane>=16) + e (this code's order, not the ISA table)
//   C/D 16x16 f32: lane N=lane%16; vgpr g -> M = g + 8*(lane>=16)

__device__ __forceinline__ v8f wmma_f16(v16h a, v16h b, v8f c) {
  return __builtin_amdgcn_wmma_f32_16x16x32_f16(false, a, false, b, (short)0, c,
                                                false, false);
}

// 2D f16 tile load: global (row-major, stride0 elems/row) -> LDS with padding.
// pad codes: interval 2^(code+1) DWORDs between pads, amount (code+1) DWORDs.
__device__ __forceinline__ void tdm_load_2d_f16(
    unsigned lds_byte_addr, const _Float16* gsrc, unsigned tensor_d0,
    unsigned tensor_d1, unsigned tile_d0, unsigned tile_d1,
    unsigned long long stride0, unsigned pad_interval_code,
    unsigned pad_amount_code) {
  unsigned long long ga = (unsigned long long)(uintptr_t)gsrc;
  u32x4 g0;
  g0[0] = 1u;                                   // count=1 (valid user desc)
  g0[1] = lds_byte_addr;                        // lds_addr [63:32]
  g0[2] = (unsigned)ga;                         // global_addr [95:64]
  g0[3] = (unsigned)((ga >> 32) & 0x01FFFFFFull) | (2u << 30);  // addr hi, type=2
  i32x8 g1;
  g1[0] = (int)((1u << 16)                      // data_size = 2 bytes
                | (1u << 20)                    // pad_enable
                | (pad_interval_code << 22) | (pad_amount_code << 25));
  g1[1] = (int)((tensor_d0 & 0xFFFFu) << 16);                  // dim0 [79:48] lo
  g1[2] = (int)((tensor_d0 >> 16) | ((tensor_d1 & 0xFFFFu) << 16));
  g1[3] = (int)((tensor_d1 >> 16) | (tile_d0 << 16));          // tile_dim0
  g1[4] = (int)tile_d1;                                        // tile_dim1 (d2=0)
  g1[5] = (int)(unsigned)stride0;                              // dim0_stride lo
  g1[6] = (int)((unsigned)(stride0 >> 32) & 0xFFFFu);
  g1[7] = 0;
#if defined(__HIP_DEVICE_COMPILE__)
  i32x4 z4 = {0, 0, 0, 0};
#if __has_include(<hip/amd_detail/amd_gfx1250_TDM.h>)
  i32x8 z8 = {0, 0, 0, 0, 0, 0, 0, 0};
  __builtin_amdgcn_tensor_load_to_lds(g0, g1, z4, z4, z8, 0);
#else
  __builtin_amdgcn_tensor_load_to_lds(g0, g1, z4, z4, 0);
#endif
#else
  (void)g0; (void)g1;  // host pass: never executed
#endif
}

// ---------------------------------------------------------------------------
// Kernel 1: QKV projection. Grid: (B*S)/64 blocks of 128 threads (4 waves).
// ---------------------------------------------------------------------------
template <bool TRANSPOSED>
__device__ __forceinline__ void proj_matrix(const v16h (&af)[8],
                                            const float* __restrict__ W,
                                            const float* __restrict__ bias,
                                            _Float16* __restrict__ Om,
                                            int rowbase, int l16, int hi) {
  for (int nt = 0; nt < 16; ++nt) {
    const int n = nt * 16 + l16;
    v8f c;
#pragma unroll
    for (int i = 0; i < 8; ++i) c[i] = 0.0f;
    const float* wrow = W + (size_t)n * D_ + hi * 16;  // B[k][n] = W[n][k]
#pragma unroll
    for (int kt = 0; kt < 8; ++kt) {
      v16h bf;
      const float* p = wrow + kt * 32;
#pragma unroll
      for (int e = 0; e < 16; ++e) bf[e] = (_Float16)p[e];
      c = wmma_f16(af[kt], bf, c);
    }
    const float bb = bias[n];
#pragma unroll
    for (int g = 0; g < 8; ++g) {
      const int fr = rowbase + g + hi * 8;  // flat row in [0, B*S)
      if (TRANSPOSED) {                     // [B][D][S]
        const int bat = fr >> 12;           // / S_
        const int s = fr & (S_ - 1);
        Om[((size_t)bat * D_ + n) * S_ + s] = (_Float16)(c[g] + bb);
      } else {                              // [B*S][D]
        Om[(size_t)fr * D_ + n] = (_Float16)(c[g] + bb);
      }
    }
  }
}

__global__ __launch_bounds__(128) void qkv_proj_kernel(
    const float* __restrict__ x, const float* __restrict__ Wq,
    const float* __restrict__ bq, const float* __restrict__ Wk,
    const float* __restrict__ bk, const float* __restrict__ Wv,
    const float* __restrict__ bv, _Float16* __restrict__ Qh,
    _Float16* __restrict__ Kh, _Float16* __restrict__ VtW) {
  const int lane = threadIdx.x & 31;
  const int wave = threadIdx.x >> 5;
  const int l16 = lane & 15;
  const int hi = lane >> 4;
  const int rowbase = blockIdx.x * 64 + wave * 16;

  // This wave's 16 x-rows as 8 A-fragments (fp32 -> f16).
  v16h af[8];
  const float* xrow = x + (size_t)(rowbase + l16) * D_;
#pragma unroll
  for (int kt = 0; kt < 8; ++kt) {
#pragma unroll
    for (int c = 0; c < 2; ++c) {
      const float* p = xrow + kt * 32 + hi * 16 + c * 8;
#pragma unroll
      for (int e = 0; e < 8; ++e) af[kt][c * 8 + e] = (_Float16)p[e];
    }
  }

  proj_matrix<false>(af, Wq, bq, Qh, rowbase, l16, hi);
  proj_matrix<false>(af, Wk, bk, Kh, rowbase, l16, hi);
  proj_matrix<true>(af, Wv, bv, VtW, rowbase, l16, hi);
}

// ---------------------------------------------------------------------------
// Kernel 2: causal flash attention. Grid (S/64, B); 128 threads (4 waves).
// Double-buffered TDM staging of 64-key K / V^T blocks.
// ---------------------------------------------------------------------------
__global__ __launch_bounds__(128) void attn_kernel(
    const _Float16* __restrict__ Qh, const _Float16* __restrict__ Kh,
    const _Float16* __restrict__ VtW, float* __restrict__ out) {
  extern __shared__ char smem_raw[];
  _Float16* KtBuf[2];
  _Float16* VtBuf[2];
  KtBuf[0] = (_Float16*)smem_raw;
  VtBuf[0] = KtBuf[0] + 64 * KT_STRIDE;
  KtBuf[1] = KtBuf[0] + SET_HALVES;
  VtBuf[1] = KtBuf[1] + 64 * KT_STRIDE;
  float* Sc_all = (float*)(KtBuf[0] + 2 * SET_HALVES);  // 4 x [16][SC_STRIDE]
  float* alphaArr = Sc_all + 4 * 16 * SC_STRIDE;        // 4 x 16
  float* lArr = alphaArr + 4 * 16;                      // 4 x 16

  const int lane = threadIdx.x & 31;
  const int wave = threadIdx.x >> 5;
  const int l16 = lane & 15;
  const int hi = lane >> 4;
  const int b = blockIdx.y;
  const int q0 = blockIdx.x * 64;
  float* Sc = Sc_all + wave * 16 * SC_STRIDE;

  // Issue both TDM tile loads for key block starting at kb2, into buffer set s.
  auto issue_block = [&](int kb2, int s) {
    // K block: 64 rows x 256 halves; pad 8dw per 128dw row -> 272-half rows.
    tdm_load_2d_f16((unsigned)(uintptr_t)KtBuf[s],
                    Kh + (size_t)(b * S_ + kb2) * D_, D_,
                    (unsigned)(S_ - kb2), 256, 64, D_, 6, 7);
    // V^T block: 256 rows x 64 halves; pad 4dw per 32dw row -> 72-half rows.
    tdm_load_2d_f16((unsigned)(uintptr_t)VtBuf[s],
                    VtW + (size_t)b * D_ * S_ + kb2,
                    (unsigned)(S_ - kb2), D_, 64, 256, S_, 4, 3);
  };

  // Preload this wave's Q rows as 8 A-fragments.
  v16h qf[8];
  {
    const _Float16* qrow = Qh + (size_t)(b * S_ + q0 + wave * 16 + l16) * D_;
#pragma unroll
    for (int kt = 0; kt < 8; ++kt) {
#pragma unroll
      for (int c = 0; c < 2; ++c) {
        const _Float16* p = qrow + kt * 32 + hi * 16 + c * 8;
#pragma unroll
        for (int e = 0; e < 8; ++e) qf[kt][c * 8 + e] = p[e];
      }
    }
  }

  v8f O[16];
#pragma unroll
  for (int dt = 0; dt < 16; ++dt)
#pragma unroll
    for (int g = 0; g < 8; ++g) O[dt][g] = 0.0f;

  const int r = lane >> 1;
  const int halfc = lane & 1;
  float m_run = -__builtin_inff();
  float l_run = 0.0f;

  const int nblocks = q0 / 64 + 1;  // causal: key blocks 0 .. q0/64
  if (wave == 0) issue_block(0, 0); // prologue: stage block 0 into set 0

  for (int j = 0; j < nblocks; ++j) {
    const int kbase = j * 64;
    const int s = j & 1;
    const _Float16* Kt = KtBuf[s];
    const _Float16* Vt = VtBuf[s];

    if (wave == 0) {
      if (j + 1 < nblocks) {
        // Prefetch next block into the other set (its readers finished at the
        // end-of-iteration barrier of j-1), then wait for THIS block's 2 TDM
        // ops (per-wave TENSORcnt, in-order) to have landed.
        issue_block((j + 1) * 64, s ^ 1);
        WAIT_TENSORCNT(2);
      } else {
        WAIT_TENSORCNT(0);
      }
    }
    __syncthreads();  // publish block j's LDS tiles to all waves

    // S = Q K^T * scale, causal-masked, into per-wave Sc (16 x 64 f32).
#pragma unroll
    for (int nt = 0; nt < 4; ++nt) {
      v8f c;
#pragma unroll
      for (int i = 0; i < 8; ++i) c[i] = 0.0f;
      const _Float16* krow = Kt + (nt * 16 + l16) * KT_STRIDE + hi * 16;
#pragma unroll
      for (int kt = 0; kt < 8; ++kt) {
        v16h bf;
        const _Float16* p = krow + kt * 32;
#pragma unroll
        for (int e = 0; e < 16; ++e) bf[e] = p[e];
        c = wmma_f16(qf[kt], bf, c);
      }
      const int kcol = kbase + nt * 16 + l16;
#pragma unroll
      for (int g = 0; g < 8; ++g) {
        const int qr = q0 + wave * 16 + g + hi * 8;
        const float sv = c[g] * SCALE_;
        Sc[(g + hi * 8) * SC_STRIDE + nt * 16 + l16] =
            (kcol > qr) ? -__builtin_inff() : sv;
      }
    }
    __syncthreads();

    // Online softmax (2 lanes per row, 32 cols each).
    {
      float* srow = Sc + r * SC_STRIDE + halfc * 32;
      float mb = -__builtin_inff();
#pragma unroll
      for (int c2 = 0; c2 < 32; ++c2) mb = fmaxf(mb, srow[c2]);
      mb = fmaxf(mb, __shfl_xor(mb, 1, 32));
      const float nm = fmaxf(m_run, mb);
      const float alpha = __expf(m_run - nm);
      float sum = 0.0f;
#pragma unroll
      for (int c2 = 0; c2 < 32; ++c2) {
        const float p = __expf(srow[c2] - nm);
        srow[c2] = p;
        sum += p;
      }
      sum += __shfl_xor(sum, 1, 32);
      l_run = l_run * alpha + sum;
      m_run = nm;
      if (halfc == 0) alphaArr[wave * 16 + r] = alpha;
    }
    __syncthreads();

    // Rescale O by per-row alpha.
    float av[8];
#pragma unroll
    for (int g = 0; g < 8; ++g) av[g] = alphaArr[wave * 16 + g + hi * 8];
#pragma unroll
    for (int dt = 0; dt < 16; ++dt)
#pragma unroll
      for (int g = 0; g < 8; ++g) O[dt][g] *= av[g];

    // P A-fragments (shared across all 16 d-tiles).
    v16h pf[2];
#pragma unroll
    for (int kt2 = 0; kt2 < 2; ++kt2) {
#pragma unroll
      for (int c = 0; c < 2; ++c) {
        const float* pr = Sc + l16 * SC_STRIDE + kt2 * 32 + hi * 16 + c * 8;
#pragma unroll
        for (int e = 0; e < 8; ++e) pf[kt2][c * 8 + e] = (_Float16)pr[e];
      }
    }

    // O += P V  (B-frag from transposed V in LDS).
#pragma unroll
    for (int dt = 0; dt < 16; ++dt) {
#pragma unroll
      for (int kt2 = 0; kt2 < 2; ++kt2) {
        v16h bf;
        const _Float16* vp =
            Vt + (dt * 16 + l16) * VT_STRIDE + kt2 * 32 + hi * 16;
#pragma unroll
        for (int e = 0; e < 16; ++e) bf[e] = vp[e];
        O[dt] = wmma_f16(pf[kt2], bf, O[dt]);
      }
    }
    __syncthreads();  // all reads of set s done before iter j+1 refills it
  }

  // Finalize.
  if (halfc == 0) lArr[wave * 16 + r] = l_run;
  __syncthreads();
  float inv[8];
#pragma unroll
  for (int g = 0; g < 8; ++g) inv[g] = 1.0f / lArr[wave * 16 + g + hi * 8];
#pragma unroll
  for (int dt = 0; dt < 16; ++dt) {
#pragma unroll
    for (int g = 0; g < 8; ++g) {
      const int qr = q0 + wave * 16 + g + hi * 8;
      out[(size_t)(b * S_ + qr) * D_ + dt * 16 + l16] = O[dt][g] * inv[g];
    }
  }
}

// ---------------------------------------------------------------------------
extern "C" void kernel_launch(void* const* d_in, const int* in_sizes, int n_in,
                              void* d_out, int out_size, void* d_ws,
                              size_t ws_size, hipStream_t stream) {
  (void)in_sizes; (void)n_in; (void)out_size; (void)ws_size;
  const float* x  = (const float*)d_in[0];
  const float* Wq = (const float*)d_in[1];
  const float* bq = (const float*)d_in[2];
  const float* Wk = (const float*)d_in[3];
  const float* bk = (const float*)d_in[4];
  const float* Wv = (const float*)d_in[5];
  const float* bv = (const float*)d_in[6];
  float* out = (float*)d_out;

  const size_t nElem = (size_t)B_ * S_ * D_;
  _Float16* Qh  = (_Float16*)d_ws;   // 8 MB
  _Float16* Kh  = Qh + nElem;        // 8 MB
  _Float16* VtW = Kh + nElem;        // 8 MB, [B][D][S]  (ws >= 24 MB)

  qkv_proj_kernel<<<dim3((B_ * S_) / 64), dim3(128), 0, stream>>>(
      x, Wq, bq, Wk, bk, Wv, bv, Qh, Kh, VtW);

  const size_t smem =
      (size_t)(2 * SET_HALVES) * sizeof(_Float16) +
      (size_t)(4 * 16 * SC_STRIDE + 64 + 64) * sizeof(float);  // ~158.5 KB
  (void)hipFuncSetAttribute((const void*)attn_kernel,
                            hipFuncAttributeMaxDynamicSharedMemorySize,
                            (int)smem);
  attn_kernel<<<dim3(S_ / 64, B_), dim3(128), smem, stream>>>(Qh, Kh, VtW, out);
}
